// ODEGRU_31490700215032
// MI455X (gfx1250) — compile-verified, matrix operand order checked by reading, not run on hardware
//
#include <hip/hip_runtime.h>
#include <hip/hip_bf16.h>

typedef __bf16 bf16;
typedef bf16  v8bf  __attribute__((ext_vector_type(8)));
typedef bf16  v16bf __attribute__((ext_vector_type(16)));
typedef float v8f   __attribute__((ext_vector_type(8)));

#define TT 128
#define DD 64
#define HHH 256      // H == HODE == 256

union ABu { v16bf v; v8bf h[2]; };

// Assemble a 16x32 bf16 WMMA A/B fragment from row-major storage.
// Lane mapping (ISA 7.12.2, 16-bit A 16x32): lanes 0-15 hold K {k0..k0+7, k0+16..k0+23},
// lanes 16-31 hold K {k0+8..k0+15, k0+24..k0+31}; row/col index = lane & 15.
__device__ __forceinline__ v16bf load_frag(const bf16* __restrict__ base, int stride,
                                           int r, int g, int k0) {
  ABu u;
  u.h[0] = *(const v8bf*)(base + r * stride + k0 + g * 8);
  u.h[1] = *(const v8bf*)(base + r * stride + k0 + 16 + g * 8);
  return u.v;
}

__device__ __forceinline__ v8f wmma_bf16(v16bf a, v16bf b, v8f c) {
  return __builtin_amdgcn_wmma_f32_16x16x32_bf16(false, a, false, b, (short)0, c,
                                                 false, false);
}

__device__ __forceinline__ v8f zero8() {
  v8f v = {0.f, 0.f, 0.f, 0.f, 0.f, 0.f, 0.f, 0.f};
  return v;
}

__device__ __forceinline__ float sigm_f(float x) {
  return 1.0f / (1.0f + __expf(-x));
}
__device__ __forceinline__ float tanh_f(float x) {
  float e = __expf(2.0f * x);
  return (e - 1.0f) / (e + 1.0f);
}

__global__ void cvt_bf16(const float* __restrict__ in, bf16* __restrict__ out, int n) {
  int i = blockIdx.x * 256 + threadIdx.x;
  if (i < n) out[i] = (bf16)in[i];
}

__global__ __launch_bounds__(512, 1) void odegru_kernel(
    const float* __restrict__ x, const float* __restrict__ times,
    const float* __restrict__ mask,
    const float* __restrict__ b1, const float* __restrict__ b2,
    const float* __restrict__ b_ih, const float* __restrict__ b_hh,
    const bf16* __restrict__ W1, const bf16* __restrict__ W2,
    const bf16* __restrict__ Wih, const bf16* __restrict__ Whh,
    float* __restrict__ out) {
  __shared__ float hS[16 * HHH];   // fp32 hidden state
  __shared__ float kS[16 * HHH];   // RK4 ksum accumulator, then fp32 h_ode
  __shared__ bf16  hA[16 * HHH];   // bf16 A-staging of h (k1 input)
  __shared__ bf16  tA[16 * HHH];   // bf16 A-staging of h + c*dt*k  / h_ode
  __shared__ bf16  zA[16 * HHH];   // bf16 A-staging of relu(... W1)
  __shared__ bf16  xA[16 * DD];    // bf16 A-staging of x_t
  __shared__ float mcol[16];       // mask column for this step

  const int tid  = threadIdx.x;
  const int wave = tid >> 5;       // 16 waves
  const int lane = tid & 31;
  const int ln   = lane & 15;      // row (A) / col (B) index within tile
  const int g    = lane >> 4;      // K sub-group
  const int m0   = blockIdx.x * 16;  // batch rows owned by this block
  const int cb   = wave * 16;        // hidden-column tile owned by this wave

  // t = 0 uses h_ode == 0
  for (int i = tid; i < 16 * HHH; i += 512) { kS[i] = 0.0f; tA[i] = (bf16)0.0f; }

  auto load_xt_mask = [&](int t) {
    for (int i = tid; i < 16 * DD; i += 512) {
      int r = i >> 6, d = i & 63;
      xA[i] = (bf16)x[(size_t)(m0 + r) * (TT * DD) + (size_t)t * DD + d];
    }
    if (tid < 16) mcol[tid] = mask[(m0 + tid) * TT + t];
  };

  // Sum of a K=256 GEMM tile against rows [nb, nb+16) of a 256-K weight matrix.
  auto gemm256 = [&](const bf16* __restrict__ A, const bf16* __restrict__ W,
                     int nb, v8f acc) -> v8f {
#pragma unroll 4
    for (int k0 = 0; k0 < HHH; k0 += 32) {
      v16bf a = load_frag(A, HHH, ln, g, k0);
      acc = wmma_bf16(a, load_frag(W, HHH, nb + ln, g, k0), acc);
    }
    return acc;
  };

  // One RK4 stage: z = relu(A@W1^T + b1); k = z@W2^T + b2; fold into ksum/tmp.
  auto rk_stage = [&](int s, float dtv) {
    const bf16* Ain = (s == 0) ? hA : tA;
    {
      v8f acc = gemm256(Ain, W1, cb, zero8());
      float b1v = b1[cb + ln];
#pragma unroll
      for (int i = 0; i < 8; ++i) {
        float v = acc[i] + b1v;
        zA[(g * 8 + i) * HHH + cb + ln] = (bf16)(v > 0.0f ? v : 0.0f);  // ReLU
      }
    }
    __syncthreads();
    {
      v8f acc = gemm256(zA, W2, cb, zero8());
      float b2v = b2[cb + ln];
#pragma unroll
      for (int i = 0; i < 8; ++i) {
        int addr  = (g * 8 + i) * HHH + cb + ln;
        float kv  = acc[i] + b2v;
        float hv  = hS[addr];
        if (s == 0) {
          kS[addr] = kv;
          tA[addr] = (bf16)(hv + 0.5f * dtv * kv);
        } else if (s == 1) {
          kS[addr] += 2.0f * kv;
          tA[addr] = (bf16)(hv + 0.5f * dtv * kv);
        } else if (s == 2) {
          kS[addr] += 2.0f * kv;
          tA[addr] = (bf16)(hv + dtv * kv);
        } else {
          float hode = hv + (dtv * (1.0f / 6.0f)) * (kS[addr] + kv);
          kS[addr] = hode;            // fp32 h_ode for the GRU blend
          tA[addr] = (bf16)hode;      // bf16 h_ode as GRU A-matrix
        }
      }
    }
    __syncthreads();
  };

  // Small K=64 GEMM against rows [nb, nb+16) of W_ih.
  auto gemm_x = [&](int nb, v8f acc) -> v8f {
#pragma unroll
    for (int k0 = 0; k0 < DD; k0 += 32) {
      v16bf a = load_frag(xA, DD, ln, g, k0);
      acc = wmma_bf16(a, load_frag(Wih, DD, nb + ln, g, k0), acc);
    }
    return acc;
  };

  // GRU update. One accumulator live at a time:
  //   r-gate and z-gate each fuse the x-GEMM and h-GEMM into one accumulator
  //   (sigmoid(i_g + h_g) distributes over the sum of the two GEMMs).
  auto gru_step = [&](int t) {
    v8f rv;                      // sigmoid(i_r + h_r)
    {
      v8f acc = gemm_x(cb, zero8());
      acc = gemm256(tA, Whh, cb, acc);
      float bb = b_ih[cb + ln] + b_hh[cb + ln];
#pragma unroll
      for (int i = 0; i < 8; ++i) rv[i] = sigm_f(acc[i] + bb);
    }
    v8f zv;                      // sigmoid(i_z + h_z)
    {
      v8f acc = gemm_x(HHH + cb, zero8());
      acc = gemm256(tA, Whh, HHH + cb, acc);
      float bb = b_ih[HHH + cb + ln] + b_hh[HHH + cb + ln];
#pragma unroll
      for (int i = 0; i < 8; ++i) zv[i] = sigm_f(acc[i] + bb);
    }
    v8f inn = gemm_x(2 * HHH + cb, zero8());             // i_n (pre-bias)
    v8f hnn = gemm256(tA, Whh, 2 * HHH + cb, zero8());   // h_n (pre-bias)
    float bin = b_ih[2 * HHH + cb + ln];
    float bhn = b_hh[2 * HHH + cb + ln];
#pragma unroll
    for (int i = 0; i < 8; ++i) {
      int mr    = g * 8 + i;
      int addr  = mr * HHH + cb + ln;
      float hode = kS[addr];  // fp32 h_ode (zero at t==0)
      float n = tanh_f((inn[i] + bin) + rv[i] * (hnn[i] + bhn));
      float hnew = (1.0f - zv[i]) * n + zv[i] * hode;
      float mv = mcol[mr];
      float ho = mv * hnew + (1.0f - mv) * hode;
      hS[addr] = ho;
      hA[addr] = (bf16)ho;
      out[(size_t)(m0 + mr) * (TT * HHH) + (size_t)t * HHH + cb + ln] = ho;
    }
  };

  load_xt_mask(0);
  __syncthreads();
  gru_step(0);

  for (int t = 1; t < TT; ++t) {
    float dtv = times[t] - times[t - 1];
    __syncthreads();            // prev-step GRU readers done before xA/mcol rewrite
    load_xt_mask(t);
    if (t + 1 < TT && tid < 16) // warm L2 for next step's x tile
      __builtin_prefetch(&x[(size_t)(m0 + tid) * (TT * DD) + (size_t)(t + 1) * DD], 0, 0);
    __syncthreads();            // xA/mcol + prev h/hA visible to all
    rk_stage(0, dtv);
    rk_stage(1, dtv);
    rk_stage(2, dtv);
    rk_stage(3, dtv);
    gru_step(t);
  }
}

extern "C" void kernel_launch(void* const* d_in, const int* in_sizes, int n_in,
                              void* d_out, int out_size, void* d_ws, size_t ws_size,
                              hipStream_t stream) {
  (void)in_sizes; (void)n_in; (void)out_size; (void)ws_size;
  const float* x     = (const float*)d_in[0];
  const float* times = (const float*)d_in[1];
  const float* maskp = (const float*)d_in[2];
  const float* W1f   = (const float*)d_in[3];
  const float* b1    = (const float*)d_in[4];
  const float* W2f   = (const float*)d_in[5];
  const float* b2    = (const float*)d_in[6];
  const float* Wihf  = (const float*)d_in[7];
  const float* b_ih  = (const float*)d_in[8];
  const float* Whhf  = (const float*)d_in[9];
  const float* b_hh  = (const float*)d_in[10];

  // bf16 weight copies in workspace (L2-resident: ~740 KB total)
  bf16* ws   = (bf16*)d_ws;
  bf16* W1b  = ws;                 // 256*256  = 65536
  bf16* W2b  = ws + 65536;         // 256*256  = 65536
  bf16* Wihb = ws + 131072;        // 768*64   = 49152
  bf16* Whhb = ws + 180224;        // 768*256  = 196608

  cvt_bf16<<<(65536 + 255) / 256, 256, 0, stream>>>(W1f, W1b, 65536);
  cvt_bf16<<<(65536 + 255) / 256, 256, 0, stream>>>(W2f, W2b, 65536);
  cvt_bf16<<<(49152 + 255) / 256, 256, 0, stream>>>(Wihf, Wihb, 49152);
  cvt_bf16<<<(196608 + 255) / 256, 256, 0, stream>>>(Whhf, Whhb, 196608);

  odegru_kernel<<<64, 512, 0, stream>>>(x, times, maskp, b1, b2, b_ih, b_hh,
                                        W1b, W2b, Wihb, Whhb, (float*)d_out);
}
